// CausalSelfAttention_77429670413056
// MI455X (gfx1250) — matrix operand order repaired and checked by reading, not compiled, not run on hardware
//
#include <hip/hip_runtime.h>

typedef __attribute__((ext_vector_type(16))) __bf16 v16bf;
typedef __attribute__((ext_vector_type(8)))  __bf16 v8bf;
typedef __attribute__((ext_vector_type(8)))  float  v8f;

#define B_  2
#define S_  2048
#define D_  1024
#define H_  16
#define HD_ 64
#define M_  (B_ * S_)   // 4096 rows

// ---- optional gfx1250 async global->LDS path (compile-safe fallback) ------
#ifndef HAVE_ASYNC_LDS
#if defined(__has_builtin)
#if __has_builtin(__builtin_amdgcn_global_load_async_to_lds_b128)
#define HAVE_ASYNC_LDS 1
#endif
#endif
#endif
#ifndef HAVE_ASYNC_LDS
#define HAVE_ASYNC_LDS 0
#endif

#if HAVE_ASYNC_LDS
// param 0 (probe-confirmed): pointer to int __attribute__((vector_size(16)))
typedef int v4i_pay __attribute__((vector_size(16)));
typedef __attribute__((address_space(3))) v4i_pay lds_v4i;
__device__ __forceinline__ void async_ld16(void* lds, const void* g) {
  __builtin_amdgcn_global_load_async_to_lds_b128((v4i_pay*)g, (lds_v4i*)lds, 0, 0);
}
__device__ __forceinline__ void wait_async0() {
#if defined(__has_builtin) && __has_builtin(__builtin_amdgcn_s_wait_asynccnt)
  __builtin_amdgcn_s_wait_asynccnt(0);
#else
  __asm__ volatile("s_wait_asynccnt 0x0" ::: "memory");
#endif
}
#endif

// ---------------------------------------------------------------- helpers

__device__ __forceinline__ __bf16 f2bf(float f) {
  union { float f; unsigned u; } a; a.f = f;
  unsigned r = (a.u + 0x7FFFu + ((a.u >> 16) & 1u)) >> 16;   // RNE
  union { unsigned short s; __bf16 b; } o; o.s = (unsigned short)r;
  return o.b;
}

__device__ __forceinline__ unsigned short bf_bits(__bf16 b) {
  union { __bf16 b; unsigned short s; } u; u.b = b; return u.s;
}

__global__ void f32_to_bf16_kern(const float* __restrict__ src,
                                 __bf16* __restrict__ dst, int n) {
  int i = blockIdx.x * blockDim.x + threadIdx.x;
  if (i < n) dst[i] = f2bf(src[i]);
}

// ---------------------------------------------------------------- GEMM
// C[M,N] = A[M,K](bf16) * B[K,N](bf16) + bias.  Tile 128x128, BK=32,
// double-buffered LDS (one barrier per K-step), 8 WMMA per wave per step.
// Wave w: wm = w&3 -> M subtiles {2wm,2wm+1}; wn = w>>2 -> N subtiles {4wn..4wn+3}.
// mode 0: fp32 row-major out.  mode 1: scatter bf16 into Q/K/V [B,H,S,HD].

#define BM 128
#define BN 128
#define BK 32

__global__ __launch_bounds__(256)
void gemm_bf16_wmma(const __bf16* __restrict__ A, const __bf16* __restrict__ Bm,
                    const float* __restrict__ bias,
                    float* __restrict__ outF,
                    __bf16* __restrict__ qd, __bf16* __restrict__ kd,
                    __bf16* __restrict__ vd,
                    int Mn, int Nn, int Kn, int mode)
{
  __shared__ alignas(64) __bf16 As[2][BM][BK];   // [m][k]
  __shared__ alignas(64) __bf16 Bt[2][BN][BK];   // [n][k] (transposed B tile)

  const int tid  = threadIdx.x;
  const int lane = tid & 31;
  const int w    = tid >> 5;
  const int wm   = w & 3;
  const int wn   = w >> 2;
  const int n0   = blockIdx.x * BN;
  const int m0   = blockIdx.y * BM;

  // staging split: A -> 32B/thread contiguous; B -> 2 rows x 8 cols, packed b32
  const int arow = tid >> 1, akk = (tid & 1) * 16;
  const int bkp  = tid & 15, bnn = (tid >> 4) * 8;

  const int nn    = lane & 15;
  const int kb8   = (lane < 16) ? 0 : 8;
  const int kb16  = (lane < 16) ? 0 : 16;
  const int mbase = (lane < 16) ? 0 : 8;

  v8f acc[2][4] = {};

  const __bf16* aptr = A + (size_t)(m0 + arow) * Kn;

  // ---- prologue: stage k0 = 0 into buffer 0
  {
#if HAVE_ASYNC_LDS
    async_ld16(&As[0][arow][akk],     aptr + akk);
    async_ld16(&As[0][arow][akk + 8], aptr + akk + 8);
#else
    v8bf a0 = *(const v8bf*)(aptr + akk);
    v8bf a1 = *(const v8bf*)(aptr + akk + 8);
    *(v8bf*)&As[0][arow][akk]     = a0;
    *(v8bf*)&As[0][arow][akk + 8] = a1;
#endif
    v8bf r0 = *(const v8bf*)(Bm + (size_t)(2 * bkp)     * Nn + n0 + bnn);
    v8bf r1 = *(const v8bf*)(Bm + (size_t)(2 * bkp + 1) * Nn + n0 + bnn);
#pragma unroll
    for (int e = 0; e < 8; ++e) {
      unsigned pk = (unsigned)bf_bits(r0[e]) | ((unsigned)bf_bits(r1[e]) << 16);
      *(unsigned*)&Bt[0][bnn + e][2 * bkp] = pk;
    }
#if HAVE_ASYNC_LDS
    wait_async0();
#endif
    __syncthreads();
  }

  int p = 0;
  for (int k0 = 0; k0 < Kn; k0 += BK) {
    const int kn = k0 + BK;
    const bool has_next = kn < Kn;

    v8bf r0 = {}, r1 = {};
#if !HAVE_ASYNC_LDS
    v8bf a0 = {}, a1 = {};
#endif
    if (has_next) {
#if HAVE_ASYNC_LDS
      async_ld16(&As[p ^ 1][arow][akk],     aptr + kn + akk);
      async_ld16(&As[p ^ 1][arow][akk + 8], aptr + kn + akk + 8);
#else
      a0 = *(const v8bf*)(aptr + kn + akk);
      a1 = *(const v8bf*)(aptr + kn + akk + 8);
#endif
      r0 = *(const v8bf*)(Bm + (size_t)(kn + 2 * bkp)     * Nn + n0 + bnn);
      r1 = *(const v8bf*)(Bm + (size_t)(kn + 2 * bkp + 1) * Nn + n0 + bnn);
      __builtin_prefetch(aptr + kn + BK + akk, 0, 1);
    }

    // ---- compute from buffer p: 2 A-frags x 4 B-frags -> 8 WMMA
    union { v16bf v; v8bf h[2]; } af[2];
#pragma unroll
    for (int mi = 0; mi < 2; ++mi) {
      const int ml = (2 * wm + mi) * 16 + nn;
      af[mi].h[0] = *(const v8bf*)&As[p][ml][kb16];
      af[mi].h[1] = *(const v8bf*)&As[p][ml][kb16 + 8];
    }
    v16bf bf[4];
#pragma unroll
    for (int ni = 0; ni < 4; ++ni) {
      const int nl = (4 * wn + ni) * 16 + nn;
      bf[ni] = *(const v16bf*)&Bt[p][nl][kb16];
    }
#pragma unroll
    for (int mi = 0; mi < 2; ++mi)
#pragma unroll
      for (int ni = 0; ni < 4; ++ni)
        acc[mi][ni] = __builtin_amdgcn_wmma_f32_16x16x32_bf16(
            false, af[mi].v, false, bf[ni], (short)0, acc[mi][ni], false, false);

    if (has_next) {
#if !HAVE_ASYNC_LDS
      *(v8bf*)&As[p ^ 1][arow][akk]     = a0;
      *(v8bf*)&As[p ^ 1][arow][akk + 8] = a1;
#endif
#pragma unroll
      for (int e = 0; e < 8; ++e) {
        unsigned pk = (unsigned)bf_bits(r0[e]) | ((unsigned)bf_bits(r1[e]) << 16);
        *(unsigned*)&Bt[p ^ 1][bnn + e][2 * bkp] = pk;
      }
#if HAVE_ASYNC_LDS
      wait_async0();
#endif
    }
    __syncthreads();
    p ^= 1;
  }

  // ---- epilogue: C layout: col n = lane&15; rows mbase+j
#pragma unroll
  for (int mi = 0; mi < 2; ++mi) {
#pragma unroll
    for (int ni = 0; ni < 4; ++ni) {
      int col = n0 + (4 * wn + ni) * 16 + nn;
      float bval = bias[col];
#pragma unroll
      for (int j = 0; j < 8; ++j) {
        int row = m0 + (2 * wm + mi) * 16 + mbase + j;
        float val = acc[mi][ni][j] + bval;
        if (mode == 0) {
          outF[(size_t)row * Nn + col] = val;
        } else {
          int bb = row >> 11, ss = row & (S_ - 1);
          int which = col >> 10, rem = col & (D_ - 1);
          int hh = rem >> 6, hd = rem & (HD_ - 1);
          __bf16* dst = (which == 0) ? qd : ((which == 1) ? kd : vd);
          dst[((((size_t)bb * H_ + hh) * S_) + ss) * HD_ + hd] = f2bf(val);
        }
      }
    }
  }
}

// ---------------------------------------------------------------- attention
// Q,K,V bf16 [B*H][S][64]. Block = one (b,h), 128 q rows (8 waves x 16).
// Flash-style online softmax, causal. ctx bf16 merged-head [B][S][D].

__global__ __launch_bounds__(256)
void attn_fwd_wmma(const __bf16* __restrict__ Q, const __bf16* __restrict__ Kk,
                   const __bf16* __restrict__ V, __bf16* __restrict__ ctx)
{
  __shared__ alignas(64) __bf16 Vt[64][32];        // [hd][key] transposed V tile
  __shared__ alignas(64) __bf16 Pl[8][16][32];     // per-wave P tile

  const int bh  = blockIdx.x >> 4;                 // S_/128 == 16 q-blocks
  const int q0  = (blockIdx.x & 15) * 128;
  const int bb  = bh >> 4;                         // H_ == 16
  const int hh  = bh & 15;
  const int tid = threadIdx.x, lane = tid & 31, w = tid >> 5;
  const int q_base = q0 + w * 16;
  const int nn    = lane & 15;
  const int mbase = (lane < 16) ? 0 : 8;
  const int kb8   = (lane < 16) ? 0 : 8;
  const int kb16  = (lane < 16) ? 0 : 16;

  const __bf16* Qb = Q  + (size_t)bh * S_ * HD_;
  const __bf16* Kb = Kk + (size_t)bh * S_ * HD_;
  const __bf16* Vb = V  + (size_t)bh * S_ * HD_;

  union fr { v16bf v; v8bf h[2]; };
  fr qf0, qf1;
  {
    const __bf16* qp = Qb + (size_t)(q_base + nn) * HD_;
    qf0.h[0] = *(const v8bf*)(qp + kb16);
    qf0.h[1] = *(const v8bf*)(qp + kb16 + 8);
    qf1.h[0] = *(const v8bf*)(qp + 32 + kb16);
    qf1.h[1] = *(const v8bf*)(qp + 32 + kb16 + 8);
  }

  float rmax[8], rsum[8];
  v8f acc[4] = {};
#pragma unroll
  for (int j = 0; j < 8; ++j) { rmax[j] = -1e30f; rsum[j] = 0.f; }

  const int vrow = tid >> 3, vhh = (tid & 7) * 8;

  for (int kt = 0; kt < q0 + 128; kt += 32) {
    __syncthreads();
    v8bf vv = *(const v8bf*)(Vb + (size_t)(kt + vrow) * HD_ + vhh);
#pragma unroll
    for (int e = 0; e < 8; ++e) Vt[vhh + e][vrow] = vv[e];
    __syncthreads();

    if (kt <= q_base + 15) {                       // wave-uniform causal skip
      v8f s[2];
#pragma unroll
      for (int g = 0; g < 2; ++g) {
        const __bf16* kp = Kb + (size_t)(kt + 16 * g + nn) * HD_;
        v16bf kf0 = *(const v16bf*)(kp + kb16);
        v16bf kf1 = *(const v16bf*)(kp + 32 + kb16);
        v8f sg = {};
        sg = __builtin_amdgcn_wmma_f32_16x16x32_bf16(false, qf0.v, false, kf0,
                                                     (short)0, sg, false, false);
        sg = __builtin_amdgcn_wmma_f32_16x16x32_bf16(false, qf1.v, false, kf1,
                                                     (short)0, sg, false, false);
        s[g] = sg;
      }

      float f[8], p0[8], p1[8];
#pragma unroll
      for (int j = 0; j < 8; ++j) {
        int row = q_base + mbase + j;
        int c0 = kt + nn, c1 = kt + 16 + nn;
        float v0 = (c0 <= row) ? s[0][j] * 0.125f : -1e30f;  // 1/sqrt(64)
        float v1 = (c1 <= row) ? s[1][j] * 0.125f : -1e30f;
        float tm = fmaxf(v0, v1);
        tm = fmaxf(tm, __shfl_xor(tm, 1, 32));
        tm = fmaxf(tm, __shfl_xor(tm, 2, 32));
        tm = fmaxf(tm, __shfl_xor(tm, 4, 32));
        tm = fmaxf(tm, __shfl_xor(tm, 8, 32));
        float nm = fmaxf(rmax[j], tm);
        f[j]  = __expf(rmax[j] - nm);
        p0[j] = __expf(v0 - nm);
        p1[j] = __expf(v1 - nm);
        float ps = p0[j] + p1[j];
        ps += __shfl_xor(ps, 1, 32);
        ps += __shfl_xor(ps, 2, 32);
        ps += __shfl_xor(ps, 4, 32);
        ps += __shfl_xor(ps, 8, 32);
        rsum[j] = rsum[j] * f[j] + ps;
        rmax[j] = nm;
      }
#pragma unroll
      for (int t = 0; t < 4; ++t)
#pragma unroll
        for (int j = 0; j < 8; ++j) acc[t][j] *= f[j];

#pragma unroll
      for (int j = 0; j < 8; ++j) {
        Pl[w][mbase + j][nn]      = f2bf(p0[j]);
        Pl[w][mbase + j][16 + nn] = f2bf(p1[j]);
      }
      __asm__ volatile("s_wait_dscnt 0x0" ::: "memory");  // intra-wave DS order

      fr pf;
      pf.h[0] = *(const v8bf*)&Pl[w][nn][kb16];
      pf.h[1] = *(const v8bf*)&Pl[w][nn][kb16 + 8];

#pragma unroll
      for (int t = 0; t < 4; ++t) {
        v16bf vf = *(const v16bf*)&Vt[t * 16 + nn][kb16];
        acc[t] = __builtin_amdgcn_wmma_f32_16x16x32_bf16(false, pf.v, false, vf,
                                                         (short)0, acc[t], false, false);
      }
    }
  }

#pragma unroll
  for (int j = 0; j < 8; ++j) {
    float inv = 1.0f / rsum[j];
    int srow = q_base + mbase + j;
    size_t base = ((size_t)bb * S_ + srow) * D_ + hh * HD_;
#pragma unroll
    for (int t = 0; t < 4; ++t)
      ctx[base + t * 16 + nn] = f2bf(acc[t][j] * inv);
  }
}

// ---------------------------------------------------------------- launch

extern "C" void kernel_launch(void* const* d_in, const int* in_sizes, int n_in,
                              void* d_out, int out_size, void* d_ws, size_t ws_size,
                              hipStream_t stream)
{
  (void)in_sizes; (void)n_in; (void)out_size; (void)ws_size;
  const float* x     = (const float*)d_in[0];
  const float* w_qkv = (const float*)d_in[1];
  const float* b_qkv = (const float*)d_in[2];
  const float* w_out = (const float*)d_in[3];
  const float* b_out = (const float*)d_in[4];
  float* out = (float*)d_out;

  char* ws = (char*)d_ws;
  __bf16* xb    = (__bf16*)(ws);                    //  8 MB  x bf16
  __bf16* wqkvb = (__bf16*)(ws + 8388608);          //  6 MB  w_qkv bf16
  __bf16* woutb = (__bf16*)(ws + 14680064);         //  2 MB  w_out bf16
  __bf16* qb    = (__bf16*)(ws + 16777216);         //  8 MB  Q [B,H,S,HD]
  __bf16* kb    = (__bf16*)(ws + 25165824);         //  8 MB  K
  __bf16* vb    = (__bf16*)(ws + 33554432);         //  8 MB  V
  __bf16* ctxb  = (__bf16*)(ws + 41943040);         //  8 MB  ctx [B,S,D]

  const int nx  = M_ * D_;
  const int nwq = D_ * 3 * D_;
  const int nwo = D_ * D_;
  f32_to_bf16_kern<<<(nx  + 255) / 256, 256, 0, stream>>>(x,     xb,    nx);
  f32_to_bf16_kern<<<(nwq + 255) / 256, 256, 0, stream>>>(w_qkv, wqkvb, nwq);
  f32_to_bf16_kern<<<(nwo + 255) / 256, 256, 0, stream>>>(w_out, woutb, nwo);

  // QKV projection: [4096,1024] x [1024,3072] -> split heads bf16
  dim3 g1(3 * D_ / BN, M_ / BM);
  gemm_bf16_wmma<<<g1, 256, 0, stream>>>(xb, wqkvb, b_qkv, nullptr,
                                         qb, kb, vb, M_, 3 * D_, D_, 1);

  // causal attention: B*H*(S/128) = 512 blocks
  attn_fwd_wmma<<<B_ * H_ * (S_ / 128), 256, 0, stream>>>(qb, kb, vb, ctxb);

  // output projection: [4096,1024] x [1024,1024] -> fp32 out + bias
  dim3 g2(D_ / BN, M_ / BM);
  gemm_bf16_wmma<<<g2, 256, 0, stream>>>(ctxb, woutb, b_out, out,
                                         nullptr, nullptr, nullptr,
                                         M_, D_, D_, 0);
}
